// DimeNetPlusPlus_7816840478894
// MI455X (gfx1250) — compile-verified, matrix operand order checked by reading, not run on hardware
//
#include <hip/hip_runtime.h>
#include <hip/hip_bf16.h>
#include <math.h>

// ---------------- model constants (match reference) ----------------
#define NRAD   6
#define NGAUSS 50
#define HID    256
#define INTE   64
#define SBF_IN 42          // NUM_SPH * NUM_RADIAL
#define CUTOFF 5.0f

typedef __attribute__((ext_vector_type(16))) __bf16 bf16x16;
typedef __attribute__((ext_vector_type(8)))  __bf16 bf16x8;
typedef __attribute__((ext_vector_type(8)))  float  f32x8;
typedef __attribute__((ext_vector_type(4)))  float  f32x4;

__device__ __forceinline__ float silu_f(float v) { return v / (1.0f + __expf(-v)); }

// ---------------------------------------------------------------------------
// Pack fp32 row-major weight [K, N] (row stride ldw) into bf16 WMMA-B-fragment
// major layout: dst[(((kt*nT + nt)*32 + lane)*16 + j)]
// lane l: n = nt*16 + l%16, half h = l/16; element j: k = kt*32 + (j<8 ? h*8+j
// : 16 + h*8 + j-8). Each lane's 16 bf16 (32B) are contiguous -> b128 loads.
// ---------------------------------------------------------------------------
__global__ void k_pack_bf16(const float* __restrict__ W, __bf16* __restrict__ dst,
                            int K, int N, int ldw) {
    int tid = blockIdx.x * blockDim.x + threadIdx.x;
    if (tid >= K * N) return;
    int j  = tid & 15;
    int l  = (tid >> 4) & 31;
    int nT = N >> 4;
    int nt = (tid >> 9) % nT;
    int kt = tid / (512 * nT);
    int h  = l >> 4;
    int n  = nt * 16 + (l & 15);
    int kk = (j < 8) ? (h * 8 + j) : (16 + h * 8 + (j - 8));
    int k  = kt * 32 + kk;
    dst[tid] = (__bf16)W[(size_t)k * ldw + n];
}

// ---------------------------------------------------------------------------
// WMMA GEMM: Y[M,N] = act(rowsel(X) @ Wp + bias) + skip
//   AT: A/X element type (__bf16 or float), YT: Y/skip element type.
//   Each wave owns a 16Mx64N strip: one A fragment register-reused across 4
//   v_wmma_f32_16x16x32_bf16 per K-step (4 accumulators). The 8 waves tile
//   (8/nq) M-tiles x nq N-quads so one block spans the full N (<=256): the A
//   matrix is streamed from HBM exactly once per GEMM.
// ---------------------------------------------------------------------------
template <typename AT, typename YT>
__global__ void __launch_bounds__(256)
k_gemm_wmma(const AT* __restrict__ X, int ldx,
            const int* __restrict__ gidx,
            const __bf16* __restrict__ Wp,
            const float* __restrict__ bias,
            const YT* __restrict__ skip,
            YT* __restrict__ Y,
            int M, int N, int K, int act) {
    const int lane = threadIdx.x & 31;
    const int wave = threadIdx.x >> 5;
    int nq = N >> 6; if (nq < 1) nq = 1; if (nq > 4) nq = 4;   // N-quads (64 cols)
    const int mw = 8 / nq;                                     // M-tiles per block
    const int qi = wave % nq;
    const int mi = wave / nq;
    const int m0 = blockIdx.x * (16 * mw) + mi * 16;
    const int nT   = N >> 4;
    const int half = lane >> 4;

    // A-operand row for this lane (32 lanes cover the same 16 rows; halves
    // differ in K). Clamp out-of-range rows; masked at store.
    int ma  = m0 + (lane & 15);
    int mac = (ma < M) ? ma : (M - 1);
    int row = gidx ? gidx[mac] : mac;
    const AT* xr = X + (size_t)row * ldx;

    f32x8 acc[4];
#pragma unroll
    for (int j = 0; j < 4; ++j) acc[j] = (f32x8){0.f,0.f,0.f,0.f,0.f,0.f,0.f,0.f};

    const int nK = K >> 5;
    for (int kt = 0; kt < nK; ++kt) {
        const int k0 = kt * 32;
        bf16x16 a;
        if constexpr (sizeof(AT) == 2) {
            bf16x8 c0 = *(const bf16x8*)(xr + k0 + half * 8);
            bf16x8 c1 = *(const bf16x8*)(xr + k0 + 16 + half * 8);
            a = __builtin_shufflevector(c0, c1, 0,1,2,3,4,5,6,7,8,9,10,11,12,13,14,15);
        } else {
            const f32x4* p = (const f32x4*)(xr + k0 + half * 8);
            const f32x4* q = (const f32x4*)(xr + k0 + 16 + half * 8);
            f32x4 a0 = p[0], a1 = p[1], a2 = q[0], a3 = q[1];
            a[0]  = (__bf16)a0[0]; a[1]  = (__bf16)a0[1]; a[2]  = (__bf16)a0[2]; a[3]  = (__bf16)a0[3];
            a[4]  = (__bf16)a1[0]; a[5]  = (__bf16)a1[1]; a[6]  = (__bf16)a1[2]; a[7]  = (__bf16)a1[3];
            a[8]  = (__bf16)a2[0]; a[9]  = (__bf16)a2[1]; a[10] = (__bf16)a2[2]; a[11] = (__bf16)a2[3];
            a[12] = (__bf16)a3[0]; a[13] = (__bf16)a3[1]; a[14] = (__bf16)a3[2]; a[15] = (__bf16)a3[3];
        }
        if (kt + 1 < nK) __builtin_prefetch(xr + k0 + 32, 0, 1);   // global_prefetch
#pragma unroll
        for (int j = 0; j < 4; ++j) {
            const int nt = qi * 4 + j;
            bf16x16 b = *(const bf16x16*)(Wp + ((((size_t)kt * nT + nt) * 32 + lane) << 4));
            acc[j] = __builtin_amdgcn_wmma_f32_16x16x32_bf16(false, a, false, b,
                                                             (short)0, acc[j], false, false);
        }
    }

    // C/D layout: VGPR r -> row m0 + (lane/16)*8 + r ; col = nt*16 + lane%16
#pragma unroll
    for (int j = 0; j < 4; ++j) {
        const int n  = (qi * 4 + j) * 16 + (lane & 15);
        const float bv = bias ? bias[n] : 0.0f;
#pragma unroll
        for (int r = 0; r < 8; ++r) {
            int m = m0 + half * 8 + r;
            if (m < M) {
                float v = acc[j][r] + bv;
                if (act) v = silu_f(v);
                if (skip) v += (float)skip[(size_t)m * N + n];
                Y[(size_t)m * N + n] = (YT)v;
            }
        }
    }
}

template <typename AT, typename YT>
static void launch_gemm(const AT* X, int ldx, const int* gidx, const __bf16* Wp,
                        const float* bias, const YT* skip, YT* Y,
                        int M, int N, int K, int act, hipStream_t stream) {
    int nq = N >> 6; if (nq < 1) nq = 1; if (nq > 4) nq = 4;
    int mw = 8 / nq;
    int rows = 16 * mw;
    dim3 gr((unsigned)((M + rows - 1) / rows));
    k_gemm_wmma<AT, YT><<<gr, 256, 0, stream>>>(X, ldx, gidx, Wp, bias, skip, Y, M, N, K, act);
}

// ---------------------------------------------------------------------------
// Edge features: Bessel RBF + envelope, 2x res(6), spin Gaussian, 2x res(50),
// concat -> lin_spin -> silu. One thread per edge.
// ---------------------------------------------------------------------------
struct EdgeParams {
    const float *rr1w[2], *rr1b[2], *rr2w[2], *rr2b[2];   // res_rbf (dim 6)
    const float *rg1w[2], *rg1b[2], *rg2w[2], *rg2b[2];   // res_g   (dim 50)
    const float *lsw, *lsb;                               // lin_spin (56 -> 6)
};

__global__ void k_edge(const float* __restrict__ dist, const float* __restrict__ mag,
                       const int* __restrict__ ii, const int* __restrict__ jj,
                       EdgeParams P, float* __restrict__ rbf, int E) {
    int e = blockIdx.x * blockDim.x + threadIdx.x;
    if (e >= E) return;
    float d   = dist[e] * (1.0f / CUTOFF);
    float p2  = d * d, p4 = p2 * p2, p5 = p4 * d, p6 = p5 * d, p7 = p6 * d;
    float env = 1.0f / d - 28.0f * p5 + 48.0f * p6 - 21.0f * p7;
    float x6[NRAD];
#pragma unroll
    for (int k = 0; k < NRAD; ++k)
        x6[k] = env * __sinf(3.14159265358979f * (float)(k + 1) * d);
    for (int rb = 0; rb < 2; ++rb) {
        const float *w1 = P.rr1w[rb], *b1 = P.rr1b[rb];
        const float *w2 = P.rr2w[rb], *b2 = P.rr2b[rb];
        float t[NRAD];
#pragma unroll
        for (int j = 0; j < NRAD; ++j) {
            float s = b1[j];
            for (int k = 0; k < NRAD; ++k) s += x6[k] * w1[k * NRAD + j];
            t[j] = silu_f(s);
        }
#pragma unroll
        for (int j = 0; j < NRAD; ++j) {
            float s = b2[j];
            for (int k = 0; k < NRAD; ++k) s += t[k] * w2[k * NRAD + j];
            x6[j] += silu_f(s);
        }
    }
    int a = ii[e], b = jj[e];
    float gd = mag[a * 3 + 0] * mag[b * 3 + 0] + mag[a * 3 + 1] * mag[b * 3 + 1] +
               mag[a * 3 + 2] * mag[b * 3 + 2];
    const float step  = CUTOFF / (float)(NGAUSS - 1);
    const float coeff = -0.5f / (step * step);
    float g[NGAUSS];
    for (int m = 0; m < NGAUSS; ++m) {
        float dd = gd - (float)m * step;
        g[m] = __expf(coeff * dd * dd);
    }
    for (int rb = 0; rb < 2; ++rb) {
        const float *w1 = P.rg1w[rb], *b1 = P.rg1b[rb];
        const float *w2 = P.rg2w[rb], *b2 = P.rg2b[rb];
        float t[NGAUSS];
        for (int j = 0; j < NGAUSS; ++j) {
            float s = b1[j];
            for (int k = 0; k < NGAUSS; ++k) s += g[k] * w1[k * NGAUSS + j];
            t[j] = silu_f(s);
        }
        for (int j = 0; j < NGAUSS; ++j) {
            float s = b2[j];
            for (int k = 0; k < NGAUSS; ++k) s += t[k] * w2[k * NGAUSS + j];
            g[j] += silu_f(s);
        }
    }
#pragma unroll
    for (int j = 0; j < NRAD; ++j) {
        float s = P.lsb[j];
        for (int k = 0; k < NRAD; ++k)   s += x6[k] * P.lsw[k * NRAD + j];
        for (int k = 0; k < NGAUSS; ++k) s += g[k] * P.lsw[(NRAD + k) * NRAD + j];
        rbf[(size_t)e * NRAD + j] = silu_f(s);
    }
}

// Node embedding table gather (to bf16) + spin one-hot linear (fp32).
__global__ void k_node(const float* __restrict__ table, const int* __restrict__ z,
                       const float* __restrict__ sfw, const float* __restrict__ sfb,
                       const int* __restrict__ s, __bf16* __restrict__ h,
                       float* __restrict__ spin, int N) {
    int t = blockIdx.x * blockDim.x + threadIdx.x;
    if (t < N * HID) {
        int n = t >> 8, c = t & 255;
        h[t] = (__bf16)table[(size_t)z[n] * HID + c];
    }
    if (t < N * 3) {
        int n = t / 3, j = t % 3;
        spin[t] = sfw[s[n] * 3 + j] + sfb[j];
    }
}

// out[e,:] = silu(rbf[e,:] @ W(6x256) + b)  (bf16 out)
__global__ void k_rbfh(const float* __restrict__ rbf, const float* __restrict__ w,
                       const float* __restrict__ b, __bf16* __restrict__ out, int E) {
    int t = blockIdx.x * blockDim.x + threadIdx.x;
    if (t >= E * HID) return;
    int e = t >> 8, n = t & 255;
    float s = b[n];
#pragma unroll
    for (int k = 0; k < NRAD; ++k) s += rbf[(size_t)e * NRAD + k] * w[k * HID + n];
    out[t] = (__bf16)silu_f(s);
}

// Finish embedding: x += spin_i@W[256:259] + spin_j@W[515:518] + bias; silu.
__global__ void k_embfin(__bf16* __restrict__ x, const float* __restrict__ spin,
                         const int* __restrict__ ii, const int* __restrict__ jj,
                         const float* __restrict__ Wemb, const float* __restrict__ bemb,
                         int E) {
    int t = blockIdx.x * blockDim.x + threadIdx.x;
    if (t >= E * HID) return;
    int e = t >> 8, n = t & 255;
    int a = ii[e], b = jj[e];
    float v = (float)x[t] + bemb[n];
#pragma unroll
    for (int j = 0; j < 3; ++j) {
        v += spin[a * 3 + j] * Wemb[(size_t)(256 + j) * HID + n];
        v += spin[b * 3 + j] * Wemb[(size_t)(515 + j) * HID + n];
    }
    x[t] = (__bf16)silu_f(v);
}

// t2[e,n] *= ((rbf[e,:] @ rbf1(6x8)) @ rbf2(8x256))[n]
__global__ void k_mulr(__bf16* __restrict__ t2, const float* __restrict__ rbf,
                       const float* __restrict__ w1, const float* __restrict__ w2, int E) {
    int t = blockIdx.x * blockDim.x + threadIdx.x;
    if (t >= E * HID) return;
    int e = t >> 8, n = t & 255;
    float r = 0.f;
#pragma unroll
    for (int m = 0; m < 8; ++m) {
        float s = 0.f;
#pragma unroll
        for (int k = 0; k < NRAD; ++k) s += rbf[(size_t)e * NRAD + k] * w1[k * 8 + m];
        r += s * w2[m * HID + n];
    }
    t2[t] = (__bf16)((float)t2[t] * r);
}

// Triplet interaction: acc64[idx_ji] += (sbf@sbf1@sbf2) * xkj64[idx_kj].
// 64 threads per triplet; sbf row + 8-dim basis staged in LDS; hw f32 atomics.
__global__ void __launch_bounds__(256)
k_trip(const float* __restrict__ sbf, const int* __restrict__ ikj,
       const int* __restrict__ iji, const float* __restrict__ w1,
       const float* __restrict__ w2, const __bf16* __restrict__ xkj,
       float* __restrict__ acc, int T) {
    __shared__ float s_sbf[4][SBF_IN];
    __shared__ float s_mid[4][8];
    int g = threadIdx.x >> 6;
    int j = threadIdx.x & 63;
    int t = blockIdx.x * 4 + g;
    bool ok = t < T;
    if (ok && j < SBF_IN) s_sbf[g][j] = sbf[(size_t)t * SBF_IN + j];
    __syncthreads();
    if (ok && j < 8) {
        float s = 0.f;
        for (int k = 0; k < SBF_IN; ++k) s += s_sbf[g][k] * w1[k * 8 + j];
        s_mid[g][j] = s;
    }
    __syncthreads();
    if (ok) {
        float s = 0.f;
#pragma unroll
        for (int m = 0; m < 8; ++m) s += s_mid[g][m] * w2[m * INTE + j];
        float v = s * (float)xkj[(size_t)ikj[t] * INTE + j];
        unsafeAtomicAdd(&acc[(size_t)iji[t] * INTE + j], v);
    }
}

// Out block, edge part: nodeacc[i_idx[e],n] += (rbf[e]@Wrbf(6x256))[n] * x[e,n]
__global__ void k_outedge(const float* __restrict__ rbf, const __bf16* __restrict__ xe,
                          const float* __restrict__ wrbf, const int* __restrict__ ii,
                          float* __restrict__ na, int E) {
    int t = blockIdx.x * blockDim.x + threadIdx.x;
    if (t >= E * HID) return;
    int e = t >> 8, n = t & 255;
    float v = 0.f;
#pragma unroll
    for (int k = 0; k < NRAD; ++k) v += rbf[(size_t)e * NRAD + k] * wrbf[k * HID + n];
    unsafeAtomicAdd(&na[(size_t)ii[e] * HID + n], v * (float)xe[t]);
}

// P[node] += X[node,:] @ wfinal(256x1)
__global__ void k_final(const float* __restrict__ X, const float* __restrict__ wf,
                        float* __restrict__ P, int N) {
    int t = blockIdx.x * blockDim.x + threadIdx.x;
    if (t >= N) return;
    float s = 0.f;
    for (int k = 0; k < HID; ++k) s += X[(size_t)t * HID + k] * wf[k];
    P[t] += s;
}

// ---------------------------------------------------------------------------
// Host orchestration.
// Param flattening assumption: d_in[0..8] = dist, magft_cat, sbf, z, s, i_idx,
// j_idx, idx_kj, idx_ji (insertion order); d_in[9..] = params leaves via jax
// tree_flatten (dict keys sorted, lists in order):
//   edge 0..17, emb 18..24, int blocks 25+24*b (24 leaves), out 97+10*o (10).
// ---------------------------------------------------------------------------
extern "C" void kernel_launch(void* const* d_in, const int* in_sizes, int n_in,
                              void* d_out, int out_size, void* d_ws, size_t ws_size,
                              hipStream_t stream) {
    (void)n_in; (void)ws_size;
    const int E = in_sizes[0];
    const int N = in_sizes[3];
    const int T = in_sizes[7];
    const float* dist = (const float*)d_in[0];
    const float* mag  = (const float*)d_in[1];
    const float* sbf  = (const float*)d_in[2];
    const int*   z    = (const int*)d_in[3];
    const int*   s    = (const int*)d_in[4];
    const int*   ii   = (const int*)d_in[5];
    const int*   jj   = (const int*)d_in[6];
    const int*   ikj  = (const int*)d_in[7];
    const int*   iji  = (const int*)d_in[8];
    auto pf = [&](int i) { return (const float*)d_in[9 + i]; };

    // ---- workspace carve-out (256B aligned) ----
    char* ws = (char*)d_ws;
    size_t off = 0;
    auto alloc = [&](size_t bytes) -> char* {
        off = (off + 255) & ~(size_t)255;
        char* p = ws + off; off += bytes; return p;
    };
    float*  rbf  = (float*)alloc((size_t)E * NRAD * 4);
    __bf16* h    = (__bf16*)alloc((size_t)N * HID * 2);
    float*  spin = (float*)alloc((size_t)N * 3 * 4);
    __bf16* x    = (__bf16*)alloc((size_t)E * HID * 2);
    __bf16* t1b  = (__bf16*)alloc((size_t)E * HID * 2);
    __bf16* t2b  = (__bf16*)alloc((size_t)E * HID * 2);
    __bf16* xk   = (__bf16*)alloc((size_t)E * INTE * 2);
    float*  ac   = (float*)alloc((size_t)E * INTE * 4);
    float*  na   = (float*)alloc((size_t)N * HID * 4);
    float*  nb   = (float*)alloc((size_t)N * HID * 4);
    __bf16* packBase = (__bf16*)alloc((size_t)4 * 1024 * 1024 * sizeof(__bf16));

    size_t poff = 0;
    auto pack = [&](const float* W, int K, int Nc, int ldw) -> const __bf16* {
        __bf16* dst = packBase + poff; poff += (size_t)K * Nc;
        int total = K * Nc;
        k_pack_bf16<<<(total + 255) / 256, 256, 0, stream>>>(W, dst, K, Nc, ldw);
        return dst;
    };
    const int EH_BLK = (E * HID + 255) / 256;

    // ---- pack all WMMA weights (bf16, fragment-major) ----
    const __bf16* pEmb0 = pack(pf(19) + (size_t)0   * HID, HID, HID, HID); // rows 0..255   (h_i)
    const __bf16* pEmb1 = pack(pf(19) + (size_t)259 * HID, HID, HID, HID); // rows 259..514 (h_j)
    const __bf16* pEmb2 = pack(pf(19) + (size_t)518 * HID, HID, HID, HID); // rows 518..773 (rbf_h)
    struct IB { const __bf16 *ji, *kj, *lin, *down, *up, *b1, *b2, *a1[2], *a2[2]; };
    IB ib[3];
    for (int bi = 0; bi < 3; ++bi) {
        int B0 = 25 + bi * 24;
        ib[bi].a1[0] = pack(pf(B0 + 1),  HID,  HID,  HID);
        ib[bi].a2[0] = pack(pf(B0 + 3),  HID,  HID,  HID);
        ib[bi].a1[1] = pack(pf(B0 + 5),  HID,  HID,  HID);
        ib[bi].a2[1] = pack(pf(B0 + 7),  HID,  HID,  HID);
        ib[bi].b1    = pack(pf(B0 + 9),  HID,  HID,  HID);
        ib[bi].b2    = pack(pf(B0 + 11), HID,  HID,  HID);
        ib[bi].down  = pack(pf(B0 + 12), HID,  INTE, INTE);
        ib[bi].ji    = pack(pf(B0 + 14), HID,  HID,  HID);
        ib[bi].kj    = pack(pf(B0 + 16), HID,  HID,  HID);
        ib[bi].lin   = pack(pf(B0 + 18), HID,  HID,  HID);
        ib[bi].up    = pack(pf(B0 + 23), INTE, HID,  HID);
    }
    struct OB { const __bf16 *up, *l[3]; };
    OB ob[4];
    for (int o = 0; o < 4; ++o) {
        int B0 = 97 + o * 10;
        ob[o].l[0] = pack(pf(B0 + 2), HID, HID, HID);
        ob[o].l[1] = pack(pf(B0 + 4), HID, HID, HID);
        ob[o].l[2] = pack(pf(B0 + 6), HID, HID, HID);
        ob[o].up   = pack(pf(B0 + 9), HID, HID, HID);
    }

    hipMemsetAsync(d_out, 0, (size_t)out_size * 4, stream);

    // ---- edge + node features ----
    EdgeParams ep;
    for (int k = 0; k < 2; ++k) {
        ep.rg1b[k] = pf(2 + 4 * k);  ep.rg1w[k] = pf(3 + 4 * k);
        ep.rg2b[k] = pf(4 + 4 * k);  ep.rg2w[k] = pf(5 + 4 * k);
        ep.rr1b[k] = pf(10 + 4 * k); ep.rr1w[k] = pf(11 + 4 * k);
        ep.rr2b[k] = pf(12 + 4 * k); ep.rr2w[k] = pf(13 + 4 * k);
    }
    ep.lsb = pf(0); ep.lsw = pf(1);
    k_edge<<<(E + 127) / 128, 128, 0, stream>>>(dist, mag, ii, jj, ep, rbf, E);
    k_node<<<(N * HID + 255) / 256, 256, 0, stream>>>(pf(24), z, pf(23), pf(22), s, h, spin, N);

    // ---- embedding block ----
    k_rbfh<<<EH_BLK, 256, 0, stream>>>(rbf, pf(21), pf(20), t1b, E);
    launch_gemm<__bf16, __bf16>(h,   HID, ii,      pEmb0, nullptr, (const __bf16*)nullptr, x, E, HID, HID, 0, stream);
    launch_gemm<__bf16, __bf16>(h,   HID, jj,      pEmb1, nullptr, x,                      x, E, HID, HID, 0, stream);
    launch_gemm<__bf16, __bf16>(t1b, HID, nullptr, pEmb2, nullptr, x,                      x, E, HID, HID, 0, stream);
    k_embfin<<<EH_BLK, 256, 0, stream>>>(x, spin, ii, jj, pf(19), pf(18), E);

    auto outblock = [&](int o, const __bf16* xe) {
        int B0 = 97 + o * 10;
        hipMemsetAsync(na, 0, (size_t)N * HID * 4, stream);
        k_outedge<<<EH_BLK, 256, 0, stream>>>(rbf, xe, pf(B0 + 7), ii, na, E);
        launch_gemm<float, float>(na, HID, nullptr, ob[o].up,   pf(B0 + 8), (const float*)nullptr, nb, N, HID, HID, 0, stream);
        launch_gemm<float, float>(nb, HID, nullptr, ob[o].l[0], pf(B0 + 1), (const float*)nullptr, na, N, HID, HID, 1, stream);
        launch_gemm<float, float>(na, HID, nullptr, ob[o].l[1], pf(B0 + 3), (const float*)nullptr, nb, N, HID, HID, 1, stream);
        launch_gemm<float, float>(nb, HID, nullptr, ob[o].l[2], pf(B0 + 5), (const float*)nullptr, na, N, HID, HID, 1, stream);
        k_final<<<(N + 255) / 256, 256, 0, stream>>>(na, pf(B0 + 0), (float*)d_out, N);
    };
    outblock(0, x);

    // ---- interaction blocks ----
    __bf16 *xc = x, *u1 = t1b, *u2 = t2b;
    for (int bi = 0; bi < 3; ++bi) {
        int B0 = 25 + bi * 24;
        launch_gemm<__bf16, __bf16>(xc, HID, nullptr, ib[bi].ji, pf(B0 + 13), (const __bf16*)nullptr, u1, E, HID, HID, 1, stream); // x_ji
        launch_gemm<__bf16, __bf16>(xc, HID, nullptr, ib[bi].kj, pf(B0 + 15), (const __bf16*)nullptr, u2, E, HID, HID, 1, stream); // x_kj
        k_mulr<<<EH_BLK, 256, 0, stream>>>(u2, rbf, pf(B0 + 19), pf(B0 + 20), E);
        launch_gemm<__bf16, __bf16>(u2, HID, nullptr, ib[bi].down, nullptr, (const __bf16*)nullptr, xk, E, INTE, HID, 1, stream);
        hipMemsetAsync(ac, 0, (size_t)E * INTE * 4, stream);
        k_trip<<<(T + 3) / 4, 256, 0, stream>>>(sbf, ikj, iji, pf(B0 + 21), pf(B0 + 22), xk, ac, T);
        launch_gemm<float, __bf16>(ac, INTE, nullptr, ib[bi].up, nullptr, u1, u2, E, HID, INTE, 1, stream);  // + x_ji
        // before residual
        launch_gemm<__bf16, __bf16>(u2, HID, nullptr, ib[bi].b1, pf(B0 + 8),  (const __bf16*)nullptr, u1, E, HID, HID, 1, stream);
        launch_gemm<__bf16, __bf16>(u1, HID, nullptr, ib[bi].b2, pf(B0 + 10), u2,                     u2, E, HID, HID, 1, stream);
        // lin + skip(x)
        launch_gemm<__bf16, __bf16>(u2, HID, nullptr, ib[bi].lin, pf(B0 + 17), xc, u1, E, HID, HID, 1, stream);
        // after residuals x2 (in-place on u1)
        for (int r = 0; r < 2; ++r) {
            int bb = B0 + r * 4;
            launch_gemm<__bf16, __bf16>(u1, HID, nullptr, ib[bi].a1[r], pf(bb + 0), (const __bf16*)nullptr, u2, E, HID, HID, 1, stream);
            launch_gemm<__bf16, __bf16>(u2, HID, nullptr, ib[bi].a2[r], pf(bb + 2), u1,                     u1, E, HID, HID, 1, stream);
        }
        __bf16* tmp = xc; xc = u1; u1 = tmp;  // new x lives in u1
        outblock(bi + 1, xc);
    }
}
